// GAT_2L_Model_55594056680046
// MI455X (gfx1250) — compile-verified, matrix operand order checked by reading, not run on hardware
//
#include <hip/hip_runtime.h>
#include <hip/hip_bf16.h>
#include <math.h>

// ---------------------------------------------------------------------------
// gfx1250 WMMA types (probe-confirmed builtin signature):
//   v8f = __builtin_amdgcn_wmma_f32_16x16x32_bf16(bool, v16bf, bool, v16bf,
//                                                 short, v8f, bool, bool)
// ---------------------------------------------------------------------------
typedef __attribute__((ext_vector_type(16))) __bf16 v16bf;
typedef __attribute__((ext_vector_type(8)))  __bf16 v8bf;
typedef __attribute__((ext_vector_type(8)))  float  v8f;

#define NEG_SLOPE 0.2f

// Load one 16-element bf16 fragment half-pair: p already offset by khalf.
// Per CDNA5 ISA 7.12.2 (16-bit A 16x32): vgpr0..3 -> K=khalf..khalf+7,
// vgpr4..7 -> K=khalf+16..khalf+23.  Two 16-byte loads (global_load_b128).
__device__ __forceinline__ v16bf load_frag(const __bf16* __restrict__ p) {
    v8bf lo = *(const v8bf*)(p);
    v8bf hi = *(const v8bf*)(p + 16);
    v16bf r;
#pragma unroll
    for (int j = 0; j < 8; ++j) { r[j] = lo[j]; r[8 + j] = hi[j]; }
    return r;
}

// ---------------------------------------------------------------------------
// GEMM: C[M,Ncol] = A[M,K] @ B[K,Ncol], bf16 operands, f32 accumulate.
//   A  : [M,K]    bf16 row-major
//   Bt : [Ncol,K] bf16 (B pre-transposed -> contiguous K per output column)
// Block = 256 threads = 8 waves; wave w -> M-subtile (blockIdx.x*8+w).
// Each wave computes 16x32 (two 16x16 accumulators sharing the A fragment).
// K is a template constant so the k-loop fully unrolls (4 or 8 WMMA pairs).
// ---------------------------------------------------------------------------
template <int K>
__global__ __launch_bounds__(256)
void gemm_wmma_bf16(const __bf16* __restrict__ A, const __bf16* __restrict__ Bt,
                    float* __restrict__ C, int M, int Ncol) {
    const int lane = threadIdx.x & 31;
    const int wid  = threadIdx.x >> 5;
    const int tm   = blockIdx.x * 8 + wid;     // 16-row tile index
    if (tm * 16 >= M) return;                  // wave-uniform early out
    const int tn   = blockIdx.y;               // 32-col tile index

    const int mrow  = tm * 16 + (lane & 15);
    const int mc    = mrow < M ? mrow : (M - 1);
    const int khalf = (lane >> 4) * 8;         // 0 (lanes 0-15) or 8 (lanes 16-31)
    const int n0    = tn * 32 + (lane & 15);
    const int n1    = n0 + 16;

    const __bf16* arow = A  + (size_t)mc * K + khalf;
    const __bf16* brow0 = Bt + (size_t)n0 * K + khalf;
    const __bf16* brow1 = Bt + (size_t)n1 * K + khalf;

    v8f acc0 = {}, acc1 = {};
#pragma unroll
    for (int k0 = 0; k0 < K; k0 += 32) {
        v16bf a  = load_frag(arow + k0);
        v16bf b0 = load_frag(brow0 + k0);
        v16bf b1 = load_frag(brow1 + k0);
        acc0 = __builtin_amdgcn_wmma_f32_16x16x32_bf16(
            false, a, false, b0, (short)0, acc0, false, false);
        acc1 = __builtin_amdgcn_wmma_f32_16x16x32_bf16(
            false, a, false, b1, (short)0, acc1, false, false);
    }

    // C/D layout: vgpr r, lane L -> row r + 8*(L>>4), col L&15.
    const int rbase = tm * 16 + (lane >> 4) * 8;
    float* crow = C + (size_t)rbase * Ncol + tn * 32 + (lane & 15);
    if (rbase + 8 <= M) {
#pragma unroll
        for (int r = 0; r < 8; ++r) {
            crow[0]  = acc0[r];
            crow[16] = acc1[r];
            crow += Ncol;
        }
    } else {
        for (int r = 0; r < 8; ++r) {
            if (rbase + r < M) { crow[0] = acc0[r]; crow[16] = acc1[r]; }
            crow += Ncol;
        }
    }
}

// ---------------------------------------------------------------------------
// Operand preparation
// ---------------------------------------------------------------------------
__global__ void f32_to_bf16(const float* __restrict__ in, __bf16* __restrict__ out,
                            int n) {
    int i = blockIdx.x * blockDim.x + threadIdx.x;
    if (i < n) out[i] = (__bf16)in[i];
}

// Wt[n*K + k] = W[k*Ncol + n]  (tiny weight matrices; one thread per element)
__global__ void transpose_to_bf16(const float* __restrict__ W, __bf16* __restrict__ Wt,
                                  int K, int Ncol) {
    int i = blockIdx.x * blockDim.x + threadIdx.x;
    if (i >= K * Ncol) return;
    int k = i / Ncol, n = i % Ncol;
    Wt[(size_t)n * K + k] = (__bf16)W[i];
}

// ---------------------------------------------------------------------------
// el[n,h] = sum_d z[n,h,d]*attn_l[h,d];  er likewise. One thread per (n,h).
// ---------------------------------------------------------------------------
__global__ void attn_scores(const float* __restrict__ z,
                            const float* __restrict__ attn_l,
                            const float* __restrict__ attn_r,
                            float* __restrict__ el, float* __restrict__ er,
                            int N, int H, int D) {
    int i = blockIdx.x * blockDim.x + threadIdx.x;
    if (i >= N * H) return;
    int n = i / H, h = i % H;
    const float* zr = z + (size_t)n * H * D + (size_t)h * D;
    const float* al = attn_l + (size_t)h * D;
    const float* ar = attn_r + (size_t)h * D;
    float sl = 0.f, sr = 0.f;
    for (int d = 0; d < D; ++d) { sl += zr[d] * al[d]; sr += zr[d] * ar[d]; }
    el[i] = sl; er[i] = sr;
}

__global__ void fill_kernel(float* __restrict__ p, float v, int n) {
    int i = blockIdx.x * blockDim.x + threadIdx.x;
    if (i < n) p[i] = v;
}

__device__ __forceinline__ float leaky(float v) {
    return v > 0.f ? v : NEG_SLOPE * v;
}

// float atomic-max via signed-max / unsigned-min on the bit pattern
__device__ __forceinline__ void atomicMaxFloat(float* addr, float val) {
    if (val >= 0.f) atomicMax((int*)addr, __float_as_int(val));
    else            atomicMin((unsigned int*)addr, (unsigned int)__float_as_int(val));
}

// Pass A: segment max of leaky_relu(el[src]+er[dst]) over dst. Thread per (e,h).
__global__ void edge_max(const float* __restrict__ el, const float* __restrict__ er,
                         const int* __restrict__ src, const int* __restrict__ dst,
                         float* __restrict__ mx, int E, int H) {
    int i = blockIdx.x * blockDim.x + threadIdx.x;
    if (i >= E * H) return;
    int e = i / H, h = i % H;
    float v = leaky(el[(size_t)src[e] * H + h] + er[(size_t)dst[e] * H + h]);
    atomicMaxFloat(&mx[(size_t)dst[e] * H + h], v);
}

// Pass B: w = exp(e - m[dst]); s[dst] += w. Thread per (e,h).
__global__ void edge_expsum(const float* __restrict__ el, const float* __restrict__ er,
                            const float* __restrict__ mx,
                            const int* __restrict__ src, const int* __restrict__ dst,
                            float* __restrict__ w, float* __restrict__ s,
                            int E, int H) {
    int i = blockIdx.x * blockDim.x + threadIdx.x;
    if (i >= E * H) return;
    int e = i / H, h = i % H;
    size_t di = (size_t)dst[e] * H + h;
    float v = leaky(el[(size_t)src[e] * H + h] + er[di]);
    float wv = __expf(v - mx[di]);
    w[i] = wv;
    atomicAdd(&s[di], wv);
}

// Pass C (heavy): acc[dst,f] += (w/s[dst,h]) * z[src,f].
// One wave32 per edge; lanes stride the F-wide feature row (coalesced).
__global__ __launch_bounds__(256)
void edge_aggregate(const float* __restrict__ z, const float* __restrict__ w,
                    const float* __restrict__ s,
                    const int* __restrict__ src, const int* __restrict__ dst,
                    float* __restrict__ acc, int E, int H, int D, int F) {
    int wave = (blockIdx.x * blockDim.x + threadIdx.x) >> 5;
    int lane = threadIdx.x & 31;
    if (wave >= E) return;
    int u = src[wave], v = dst[wave];
    const float* zr = z + (size_t)u * F;
    float*       ar = acc + (size_t)v * F;
    for (int f = lane; f < F; f += 32) {
        int h = f / D;
        float alpha = w[(size_t)wave * H + h] / s[(size_t)v * H + h];
        atomicAdd(&ar[f], alpha * zr[f]);
    }
}

// h_bf16 = bf16(relu(acc + b))  — layer-2 GEMM consumes bf16 directly.
__global__ void bias_relu_bf16(const float* __restrict__ acc, const float* __restrict__ b,
                               __bf16* __restrict__ hb, int N, int F) {
    int i = blockIdx.x * blockDim.x + threadIdx.x;
    if (i >= N * F) return;
    float v = acc[i] + b[i % F];
    hb[i] = (__bf16)(v > 0.f ? v : 0.f);
}

__global__ void bias_sigmoid(const float* __restrict__ acc, const float* __restrict__ b,
                             float* __restrict__ out, int N, int F) {
    int i = blockIdx.x * blockDim.x + threadIdx.x;
    if (i >= N * F) return;
    float v = acc[i] + b[i % F];
    out[i] = 1.f / (1.f + __expf(-v));
}

// ---------------------------------------------------------------------------
// Host-side orchestration (single stream, graph-capture safe: kernels only).
// ---------------------------------------------------------------------------
extern "C" void kernel_launch(void* const* d_in, const int* in_sizes, int n_in,
                              void* d_out, int out_size, void* d_ws, size_t ws_size,
                              hipStream_t stream) {
    const float* x   = (const float*)d_in[0];
    const int*   src = (const int*)  d_in[1];
    const int*   dst = (const int*)  d_in[2];
    const float* W1  = (const float*)d_in[3];
    const float* al1 = (const float*)d_in[4];
    const float* ar1 = (const float*)d_in[5];
    const float* b1  = (const float*)d_in[6];
    const float* W2  = (const float*)d_in[7];
    const float* al2 = (const float*)d_in[8];
    const float* ar2 = (const float*)d_in[9];
    const float* b2  = (const float*)d_in[10];
    float* out = (float*)d_out;

    const int FIN = 128, H = 4, D1 = 64, F1 = 256, D2 = 8, F2 = 32;
    const int N = in_sizes[0] / FIN;
    const int E = in_sizes[1];

    // ---- Workspace layout: f32 region first, then bf16 region (16B-aligned
    //      offsets: every sub-buffer size is a multiple of 16 bytes). ----
    float* ws = (float*)d_ws;
    size_t off = 0;
    float* z    = ws + off; off += (size_t)N * F1;   // z1, later z2
    float* hacc = ws + off; off += (size_t)N * F1;   // layer-1 f32 accumulator
    float* wbuf = ws + off; off += (size_t)E * H;    // per-edge softmax weights
    float* el   = ws + off; off += (size_t)N * H;
    float* er   = ws + off; off += (size_t)N * H;
    float* mx   = ws + off; off += (size_t)N * H;
    float* sm   = ws + off; off += (size_t)N * H;
    float* acc2 = ws + off; off += (size_t)N * F2;

    __bf16* bb = (__bf16*)(ws + off);
    size_t boff = 0;
    __bf16* xb  = bb + boff; boff += (size_t)N * FIN;   // x in bf16
    __bf16* hb  = bb + boff; boff += (size_t)N * F1;    // relu(h) in bf16
    __bf16* W1t = bb + boff; boff += (size_t)F1 * FIN;  // W1^T bf16 [256,128]
    __bf16* W2t = bb + boff; boff += (size_t)F2 * F1;   // W2^T bf16 [32,256]

    const int TB = 256;
    const float NEG_INF = -3.4e38f;
    dim3 blk(TB);
    auto g1 = [&](int n) { return dim3((unsigned)((n + TB - 1) / TB)); };

    // -------- Operand prep + layer-1 init --------
    f32_to_bf16<<<g1(N * FIN), blk, 0, stream>>>(x, xb, N * FIN);
    transpose_to_bf16<<<g1(FIN * F1), blk, 0, stream>>>(W1, W1t, FIN, F1);
    transpose_to_bf16<<<g1(F1 * F2), blk, 0, stream>>>(W2, W2t, F1, F2);
    fill_kernel<<<g1(N * F1), blk, 0, stream>>>(hacc, 0.f, N * F1);
    fill_kernel<<<g1(N * H),  blk, 0, stream>>>(mx, NEG_INF, N * H);
    fill_kernel<<<g1(N * H),  blk, 0, stream>>>(sm, 0.f, N * H);

    // -------- Layer 1 --------
    gemm_wmma_bf16<128><<<dim3((unsigned)((N + 127) / 128), (unsigned)(F1 / 32)),
                          blk, 0, stream>>>(xb, W1t, z, N, F1);

    attn_scores<<<g1(N * H), blk, 0, stream>>>(z, al1, ar1, el, er, N, H, D1);
    edge_max   <<<g1(E * H), blk, 0, stream>>>(el, er, src, dst, mx, E, H);
    edge_expsum<<<g1(E * H), blk, 0, stream>>>(el, er, mx, src, dst, wbuf, sm, E, H);
    edge_aggregate<<<dim3((unsigned)((E + 7) / 8)), blk, 0, stream>>>(
        z, wbuf, sm, src, dst, hacc, E, H, D1, F1);
    bias_relu_bf16<<<g1(N * F1), blk, 0, stream>>>(hacc, b1, hb, N, F1);

    // -------- Layer 2 --------
    fill_kernel<<<g1(N * F2), blk, 0, stream>>>(acc2, 0.f, N * F2);
    fill_kernel<<<g1(N * H),  blk, 0, stream>>>(mx, NEG_INF, N * H);
    fill_kernel<<<g1(N * H),  blk, 0, stream>>>(sm, 0.f, N * H);

    gemm_wmma_bf16<256><<<dim3((unsigned)((N + 127) / 128), (unsigned)(F2 / 32)),
                          blk, 0, stream>>>(hb, W2t, z, N, F2);  // z = z2 [N,32]

    attn_scores<<<g1(N * H), blk, 0, stream>>>(z, al2, ar2, el, er, N, H, D2);
    edge_max   <<<g1(E * H), blk, 0, stream>>>(el, er, src, dst, mx, E, H);
    edge_expsum<<<g1(E * H), blk, 0, stream>>>(el, er, mx, src, dst, wbuf, sm, E, H);
    edge_aggregate<<<dim3((unsigned)((E + 7) / 8)), blk, 0, stream>>>(
        z, wbuf, sm, src, dst, acc2, E, H, D2, F2);
    bias_sigmoid<<<g1(N * F2), blk, 0, stream>>>(acc2, b2, out, N, F2);
}
